// Codec_85950885527806
// MI455X (gfx1250) — matrix operand order repaired and checked by reading, not compiled, not run on hardware
//
#include <hip/hip_runtime.h>
#include <hip/hip_bf16.h>
#include <math.h>

typedef __bf16 bf16_t;
typedef __attribute__((ext_vector_type(16))) __bf16 v16bf;
typedef __attribute__((ext_vector_type(8)))  float  v8f;
typedef int v4i __attribute__((vector_size(4 * sizeof(int))));

#define AS_GLOBAL __attribute__((address_space(1)))
#define AS_LDS    __attribute__((address_space(3)))

#define IMG_B   16
#define IMG_H   512
#define IMG_W   512
#define NCHAN   16
#define TILE_W  128
#define PLANE   (IMG_H * IMG_W)          /* 262144 */
#define TOTAL   (IMG_B * 3 * PLANE)      /* 12582912 */
#define NBLK_RED 2048
#define LFRAG   (5 * 32 * 16)            /* bf16 elems of one layer's A-fragments */

#if defined(__has_builtin)
#  if __has_builtin(__builtin_amdgcn_global_load_async_to_lds_b128)
#    define HAVE_ASYNC_LDS 1
#  endif
#  if __has_builtin(__builtin_amdgcn_s_wait_asynccnt)
#    define HAVE_WAIT_ASYNC 1
#  endif
#endif

// ---------------------------------------------------------------------------
// Pack per-layer conv weights (fp32 OIHW, zero-padded to 16x16x3x3) into the
// exact per-lane WMMA A-fragment layout: for each K-step (K=32 = 2 taps x 16
// cin), lane l holds M = l&15, and element j maps to
//   K = (l>=16?16:0) + j   (contiguous order, matches B staging)
// Global K index = step*32 + K ;  tap = K/16 (0..8, 9 = zero pad), cin = K%16.
// ---------------------------------------------------------------------------
__global__ __launch_bounds__(160)
void prep_weights(const float* w0, const float* w1, const float* w2, const float* w3,
                  const float* w4, const float* w5, const float* w6, const float* w7,
                  bf16_t* __restrict__ wfrag) {
  int l    = blockIdx.x;
  int t    = threadIdx.x;        // 0..159
  int step = t >> 5;
  int lane = t & 31;
  int m    = lane & 15;
  int half = lane >> 4;

  const float* w; int cin; int cout;
  switch (l) {
    case 0: w = w0; cin = 2;  cout = 16; break;
    case 1: w = w1; cin = 16; cout = 16; break;
    case 2: w = w2; cin = 16; cout = 16; break;
    case 3: w = w3; cin = 16; cout = 16; break;
    case 4: w = w4; cin = 16; cout = 16; break;
    case 5: w = w5; cin = 16; cout = 16; break;
    case 6: w = w6; cin = 16; cout = 16; break;
    default: w = w7; cin = 16; cout = 3; break;
  }

  bf16_t* dst = wfrag + ((size_t)(l * 5 + step) * 32 + lane) * 16;
#pragma unroll
  for (int j = 0; j < 16; ++j) {
    int kk = half * 16 + j;
    int kg = step * 32 + kk;
    int tap = kg >> 4;
    int ci  = kg & 15;
    float v = 0.0f;
    if (tap < 9 && m < cout && ci < cin)
      v = w[((m * cin + ci) * 3 + tap / 3) * 3 + (tap % 3)];
    dst[j] = (bf16_t)v;
  }
}

// ---------------------------------------------------------------------------
// Build the 16-channel bf16 activation tensor [B][H][W][16] from two fp32
// planes (channels c0/c1 of a (B,3,H,W) tensor); channels 2..15 = 0.
// ---------------------------------------------------------------------------
__global__ __launch_bounds__(256)
void pack_input(const float* __restrict__ s0, int c0,
                const float* __restrict__ s1, int c1,
                bf16_t* __restrict__ act) {
  int i = blockIdx.x * 256 + threadIdx.x;        // pixel over B*H*W
  if (i >= IMG_B * PLANE) return;
  int b = i / PLANE;
  int e = i - b * PLANE;
  union { bf16_t h[16]; uint4 q[2]; } u;
#pragma unroll
  for (int c = 0; c < 16; ++c) u.h[c] = (bf16_t)0.0f;
  u.h[0] = (bf16_t)s0[((size_t)(b * 3 + c0) << 18) + e];
  u.h[1] = (bf16_t)s1[((size_t)(b * 3 + c1) << 18) + e];
  uint4* d = (uint4*)act + (size_t)i * 2;
  d[0] = u.q[0];
  d[1] = u.q[1];
}

// ---------------------------------------------------------------------------
// 3x3 SAME conv, 16->16 channels, implicit GEMM on v_wmma_f32_16x16x32_bf16.
// Block = 256 threads = 8 waves; each wave owns a 16-pixel x 16-channel tile.
// Halo tile is staged into LDS with GLOBAL_LOAD_ASYNC_TO_LDS_B128 (ASYNCcnt)
// when available, bypassing the register file.
// mode 0: out = leaky(conv + bias)  (bf16 activations)
// mode 1: clip(conv+bias) channels 0..2 -> median3 -> res = x[chan] - median
// ---------------------------------------------------------------------------
__global__ __launch_bounds__(256)
void conv3x3(const bf16_t* __restrict__ in, const bf16_t* __restrict__ wfrag,
             const float* __restrict__ bias, bf16_t* __restrict__ out,
             int mode, const float* __restrict__ xfull, int chan,
             float* __restrict__ resfull) {
  // 3 halo rows x (TILE_W+2) cols x 16 channels, plus one all-zero chunk for
  // the K-padding tap (tap 9).
  __shared__ v16bf smem[3 * (TILE_W + 2) + 1];

  int bimg = blockIdx.z;
  int y    = blockIdx.y;
  int x0   = blockIdx.x * TILE_W;
  int tid  = threadIdx.x;

  __builtin_prefetch(wfrag, 0, 1);   // global_prefetch_b8 on the A-fragments

  // ---- cooperative halo-tile load (each (row,col) chunk = 32B = 2x uint4)
  uint4* s4 = (uint4*)smem;
  const uint4* g4 = (const uint4*)in;
#if defined(HAVE_ASYNC_LDS)
  for (int i = tid; i < (3 * (TILE_W + 2) + 1) * 2; i += 256) {
    int chunk = i >> 1;
    int hs    = i & 1;
    int r     = chunk / (TILE_W + 2);
    int cc    = chunk - r * (TILE_W + 2);
    int gy    = y + r - 1;
    int gx    = x0 + cc - 1;
    bool inr  = (chunk < 3 * (TILE_W + 2)) &&
                gy >= 0 && gy < IMG_H && gx >= 0 && gx < IMG_W;
    if (inr) {
      size_t gi = ((((size_t)bimg * IMG_H + gy) * IMG_W + gx) * 2 + hs);
      __builtin_amdgcn_global_load_async_to_lds_b128(
          (AS_GLOBAL v4i*)(g4 + gi),
          (AS_LDS v4i*)(s4 + i),
          0, 0);
    } else {
      s4[i] = make_uint4(0u, 0u, 0u, 0u);
    }
  }
#  if defined(HAVE_WAIT_ASYNC)
  __builtin_amdgcn_s_wait_asynccnt(0);
#  else
  asm volatile("s_wait_asynccnt 0x0" ::: "memory");
#  endif
#else
  for (int i = tid; i < (3 * (TILE_W + 2) + 1) * 2; i += 256) {
    int chunk = i >> 1;
    int hs    = i & 1;
    uint4 val = make_uint4(0u, 0u, 0u, 0u);
    if (chunk < 3 * (TILE_W + 2)) {
      int r  = chunk / (TILE_W + 2);
      int cc = chunk - r * (TILE_W + 2);
      int gy = y + r - 1;
      int gx = x0 + cc - 1;
      if (gy >= 0 && gy < IMG_H && gx >= 0 && gx < IMG_W) {
        size_t gi = (((size_t)bimg * IMG_H + gy) * IMG_W + gx);
        val = g4[gi * 2 + hs];
      }
    }
    s4[i] = val;
  }
#endif
  __syncthreads();

  int lane = tid & 31;
  int wv   = tid >> 5;
  int n    = lane & 15;
  int half = lane >> 4;
  int px   = wv * 16 + n;            // pixel within the 128-wide strip

  const v16bf* wf = (const v16bf*)wfrag;
  v8f acc = {};

#pragma unroll
  for (int step = 0; step < 5; ++step) {
    v16bf a = wf[step * 32 + lane];                  // A: weights fragment
    int tap = step * 2 + half;                       // lanes 0-15: tap 2k, 16-31: 2k+1
    int ci  = (tap < 9) ? (tap / 3) * (TILE_W + 2) + px + (tap % 3)
                        : 3 * (TILE_W + 2);          // zero chunk (K padding)
    v16bf b = smem[ci];                              // B: 16 cin for (pixel,tap)
    acc = __builtin_amdgcn_wmma_f32_16x16x32_bf16(
        /*neg_a=*/false, a, /*neg_b=*/false, b,
        /*c_mod=*/(short)0, acc, /*reuse_a=*/false, /*reuse_b=*/false);
  }

  if (mode == 0) {
    union { bf16_t h[8]; uint4 q; } o;
#pragma unroll
    for (int v = 0; v < 8; ++v) {
      int m = half * 8 + v;
      float val = acc[v] + bias[m];
      val = (val >= 0.0f) ? val : 0.01f * val;       // leaky relu
      o.h[v] = (bf16_t)val;
    }
    size_t gi = (((size_t)bimg * IMG_H + y) * IMG_W + (x0 + px));
    ((uint4*)out)[gi * 2 + half] = o.q;
  } else {
    if (half == 0) {                                  // channels 0..2 live here
      float a0 = fminf(fmaxf(acc[0] + bias[0], -1.0f), 1.0f);
      float a1 = fminf(fmaxf(acc[1] + bias[1], -1.0f), 1.0f);
      float a2 = fminf(fmaxf(acc[2] + bias[2], -1.0f), 1.0f);
      float med = fmaxf(fminf(a0, fmaxf(a1, a2)), fminf(a1, a2));
      size_t pi = ((size_t)(bimg * 3 + chan) << 18) + (size_t)y * IMG_W + (x0 + px);
      resfull[pi] = xfull[pi] - med;
    }
  }
}

// ---------------------------------------------------------------------------
// clamp-grad delta at pixel e of a (H,W) plane (zero pad at top/left)
// ---------------------------------------------------------------------------
__device__ inline float delta_at(const float* __restrict__ p, int e) {
  int y = e >> 9;
  int x = e & (IMG_W - 1);
  float v  = p[e];
  float n  = (y > 0) ? p[e - IMG_W] : 0.0f;
  float w  = (x > 0) ? p[e - 1] : 0.0f;
  float nw = (y > 0 && x > 0) ? p[e - IMG_W - 1] : 0.0f;
  float g  = n + w - nw;
  float lo = fminf(n, w), hi = fmaxf(n, w);
  return v - fminf(fmaxf(g, lo), hi);
}

// ---------------------------------------------------------------------------
// Stage-1 sum of squares (grid-stride, fixed order -> deterministic)
// ---------------------------------------------------------------------------
__global__ __launch_bounds__(256)
void sq_partial(const float* __restrict__ src, int use_delta, float* __restrict__ partial) {
  __shared__ float red[256];
  double s = 0.0;
  for (long long f = (long long)blockIdx.x * 256 + threadIdx.x; f < TOTAL;
       f += (long long)NBLK_RED * 256) {
    float d;
    if (!use_delta) {
      d = src[f];
    } else {
      int plane = (int)(f >> 18);
      int e     = (int)(f & (PLANE - 1));
      d = delta_at(src + ((size_t)plane << 18), e);
    }
    s += (double)d * (double)d;
  }
  red[threadIdx.x] = (float)s;
  __syncthreads();
  for (int t = 128; t > 0; t >>= 1) {
    if (threadIdx.x < t) red[threadIdx.x] += red[threadIdx.x + t];
    __syncthreads();
  }
  if (threadIdx.x == 0) partial[blockIdx.x] = red[0];
}

// ---------------------------------------------------------------------------
// Per-(batch,channel) 256-bin histogram + entropy (one block per channel)
// ---------------------------------------------------------------------------
__global__ __launch_bounds__(256)
void hist_ent(const float* __restrict__ src, int use_delta, float* __restrict__ ent_out) {
  __shared__ unsigned int hist[256];
  __shared__ float red[256];
  int ch = blockIdx.x;                       // 0..47
  hist[threadIdx.x] = 0u;
  __syncthreads();
  const float* p = src + ((size_t)ch << 18);
  for (int e = threadIdx.x; e < PLANE; e += 256) {
    float v = use_delta ? delta_at(p, e) : p[e];
    if (v >= -1.0f && v <= 1.0f) {
      int idx = (int)floorf((v + 1.0f) * 128.0f);
      idx = idx < 0 ? 0 : (idx > 255 ? 255 : idx);
      atomicAdd(&hist[idx], 1u);
    }
  }
  __syncthreads();
  float pr = (float)hist[threadIdx.x] * (1.0f / (float)PLANE);
  red[threadIdx.x] = (pr > 0.0f) ? -pr * log2f(pr) : 0.0f;
  __syncthreads();
  for (int t = 128; t > 0; t >>= 1) {
    if (threadIdx.x < t) red[threadIdx.x] += red[threadIdx.x + t];
    __syncthreads();
  }
  if (threadIdx.x == 0) ent_out[ch] = red[0];
}

// ---------------------------------------------------------------------------
// Final: 4 scalars -> d_out  (loss1, loss0, invcr0, invcr1)
// ---------------------------------------------------------------------------
__global__ __launch_bounds__(256)
void finalize_k(const float* __restrict__ part0, const float* __restrict__ part1,
                const float* __restrict__ entx, const float* __restrict__ entd,
                float* __restrict__ out) {
  __shared__ float red[256];
  for (int which = 0; which < 4; ++which) {
    const float* src = (which == 0) ? part1 : (which == 1) ? part0
                       : (which == 2) ? entx : entd;
    int cnt = (which < 2) ? NBLK_RED : 48;
    float s = 0.0f;
    for (int i = threadIdx.x; i < cnt; i += 256) s += src[i];
    red[threadIdx.x] = s;
    __syncthreads();
    for (int t = 128; t > 0; t >>= 1) {
      if (threadIdx.x < t) red[threadIdx.x] += red[threadIdx.x + t];
      __syncthreads();
    }
    if (threadIdx.x == 0) {
      if (which < 2) out[which] = 128.0f * sqrtf(red[0] / (float)TOTAL);
      else           out[which] = red[0] / (8.0f * 48.0f);
    }
    __syncthreads();
  }
}

// ---------------------------------------------------------------------------
extern "C" void kernel_launch(void* const* d_in, const int* in_sizes, int n_in,
                              void* d_out, int out_size, void* d_ws, size_t ws_size,
                              hipStream_t stream) {
  const float* x = (const float*)d_in[0];
  const float* w[8];
  const float* bia[8];
  for (int l = 0; l < 8; ++l) {
    w[l]   = (const float*)d_in[1 + 2 * l];
    bia[l] = (const float*)d_in[2 + 2 * l];
  }

  char* base = (char*)d_ws;
  const size_t ACT_BYTES = (size_t)IMG_B * IMG_H * IMG_W * NCHAN * sizeof(bf16_t); // 128 MiB
  const size_t RES_BYTES = (size_t)IMG_B * 3 * IMG_H * IMG_W * sizeof(float);      // 48 MiB
  bf16_t* actA  = (bf16_t*)base;
  bf16_t* actB  = (bf16_t*)(base + ACT_BYTES);
  float*  res   = (float*)(base + 2 * ACT_BYTES);
  bf16_t* wfrag = (bf16_t*)(base + 2 * ACT_BYTES + RES_BYTES);
  float*  part0 = (float*)(base + 2 * ACT_BYTES + RES_BYTES + 65536);
  float*  part1 = part0 + NBLK_RED;
  float*  entx  = part1 + NBLK_RED;
  float*  entd  = entx + 64;

  prep_weights<<<8, 160, 0, stream>>>(w[0], w[1], w[2], w[3], w[4], w[5], w[6], w[7], wfrag);

  dim3 cgrid(IMG_W / TILE_W, IMG_H, IMG_B);
  for (int p = 0; p < 3; ++p) {
    const float* s0; int c0; const float* s1; int c1;
    if (p == 0)      { s0 = x;   c0 = 1; s1 = x;   c1 = 2; }  // predict r from (g,b)
    else if (p == 1) { s0 = res; c0 = 0; s1 = x;   c1 = 2; }  // predict g from (r',b)
    else             { s0 = res; c0 = 0; s1 = res; c1 = 1; }  // predict b from (r',g')
    pack_input<<<(IMG_B * PLANE + 255) / 256, 256, 0, stream>>>(s0, c0, s1, c1, actA);

    bf16_t* cur = actA;
    bf16_t* nxt = actB;
    for (int l = 0; l < 7; ++l) {
      conv3x3<<<cgrid, 256, 0, stream>>>(cur, wfrag + (size_t)l * LFRAG, bia[l], nxt,
                                         0, nullptr, 0, nullptr);
      bf16_t* t = cur; cur = nxt; nxt = t;
    }
    conv3x3<<<cgrid, 256, 0, stream>>>(cur, wfrag + (size_t)7 * LFRAG, bia[7], nxt,
                                       1, x, p, res);
  }

  sq_partial<<<NBLK_RED, 256, 0, stream>>>(x, 0, part0);
  sq_partial<<<NBLK_RED, 256, 0, stream>>>(res, 1, part1);
  hist_ent<<<48, 256, 0, stream>>>(x, 0, entx);
  hist_ent<<<48, 256, 0, stream>>>(res, 1, entd);
  finalize_k<<<1, 256, 0, stream>>>(part0, part1, entx, entd, (float*)d_out);
}
